// GraLSP_75866302317037
// MI455X (gfx1250) — compile-verified, matrix operand order checked by reading, not run on hardware
//
#include <hip/hip_runtime.h>
#include <hip/hip_bf16.h>
#include <math.h>

// ---------------------------------------------------------------------------
// Problem constants (match reference)
// ---------------------------------------------------------------------------
#define NNODES  100000
#define FDIM    128
#define HDIM    256
#define EDIM    128
#define KNB     20
#define WDIM    30
#define NWALK   100
#define BATCH   1024
#define NEGS    512
#define S0TOT   (BATCH + BATCH + NEGS)   // 2560 combined rows (keys|labels|negs)
#define S1TOT   (S0TOT * KNB)            // 51200 hop-1 rows

typedef __bf16 bf16;
typedef bf16  bf16x4  __attribute__((ext_vector_type(4)));
typedef bf16  bf16x16 __attribute__((ext_vector_type(16)));
typedef float floatx8 __attribute__((ext_vector_type(8)));

// ---------------------------------------------------------------------------
// Workspace layout (bytes). Total ~59.7 MB.
// ---------------------------------------------------------------------------
#define ALIGN256(x) (((x) + 255) & ~(size_t)255)
static const size_t OFF_S0   = 0;                                      // 2560 i32
static const size_t OFF_S1   = ALIGN256(OFF_S0   + S0TOT * 4);         // 51200 i32
static const size_t OFF_AMP1 = ALIGN256(OFF_S1   + S1TOT * 4);         // 100x128 f32
static const size_t OFF_AMP2 = ALIGN256(OFF_AMP1 + NWALK * FDIM * 4);  // 100x256 f32
static const size_t OFF_BP1  = ALIGN256(OFF_AMP2 + NWALK * HDIM * 4);  // 256x256 bf16 packed
static const size_t OFF_BP2  = ALIGN256(OFF_BP1  + 256 * 256 * 2);     // 512x128 bf16 packed
static const size_t OFF_XA1  = ALIGN256(OFF_BP2  + 512 * 128 * 2);     // [51200,256] bf16
static const size_t OFF_XA0  = ALIGN256(OFF_XA1  + (size_t)S1TOT * 256 * 2);
static const size_t OFF_H1   = ALIGN256(OFF_XA0  + (size_t)S0TOT * 256 * 2);
static const size_t OFF_H0   = ALIGN256(OFF_H1   + (size_t)S1TOT * 256 * 2);
static const size_t OFF_XA2  = ALIGN256(OFF_H0   + (size_t)S0TOT * 256 * 2);
static const size_t OFF_OUT2 = ALIGN256(OFF_XA2  + (size_t)S0TOT * 512 * 2);

// ---------------------------------------------------------------------------
// 1) Precompute amp tables: AMP1[p,f] = sigmoid(we[p]@wp1 + bp1),
//                           AMP2[p,h] = sigmoid(we[p]@wp2 + bp2)
//    (only 100 distinct walk types -> kills the per-neighbor sigmoid GEMM)
// ---------------------------------------------------------------------------
__global__ __launch_bounds__(384) void amp_kernel(
    const float* __restrict__ we, const float* __restrict__ wp1,
    const float* __restrict__ bp1, const float* __restrict__ wp2,
    const float* __restrict__ bp2, float* __restrict__ amp1,
    float* __restrict__ amp2) {
  const int p = blockIdx.x;
  const int t = threadIdx.x;
  const float* w = we + p * WDIM;
  if (t < FDIM) {
    float s = bp1[t];
#pragma unroll
    for (int d = 0; d < WDIM; ++d) s = fmaf(w[d], wp1[d * FDIM + t], s);
    amp1[p * FDIM + t] = 1.0f / (1.0f + __expf(-s));
  } else {
    const int f = t - FDIM;
    float s = bp2[f];
#pragma unroll
    for (int d = 0; d < WDIM; ++d) s = fmaf(w[d], wp2[d * HDIM + f], s);
    amp2[p * HDIM + f] = 1.0f / (1.0f + __expf(-s));
  }
}

// ---------------------------------------------------------------------------
// 2) Pack stacked weights [Wn ; Ws] (Kdim x N, f32) into per-lane-contiguous
//    bf16 WMMA B-fragment layout: Bp[nt][kt][lane][8 pairs] (32 B per lane).
// ---------------------------------------------------------------------------
__global__ __launch_bounds__(256) void pack_w_kernel(
    const float* __restrict__ Wn, const float* __restrict__ Ws,
    bf16* __restrict__ Bp, int Khalf, int N) {
  const int total = 2 * Khalf * N;
  const int idx = blockIdx.x * blockDim.x + threadIdx.x;
  if (idx >= total) return;
  const int nkt  = (2 * Khalf) >> 5;
  const int j2   = idx & 15;
  const int lane = (idx >> 4) & 31;
  const int rest = idx >> 9;
  const int kt   = rest % nkt;
  const int nt   = rest / nkt;
  const int n    = nt * 16 + (lane & 15);
  const int half = lane >> 4;
  const int j    = j2 >> 1;
  const int e    = j2 & 1;
  const int kk   = 2 * j + (j >= 4 ? 8 : 0) + 8 * half + e;
  const int k    = kt * 32 + kk;
  const float v  = (k < Khalf) ? Wn[(size_t)k * N + n] : Ws[(size_t)(k - Khalf) * N + n];
  Bp[idx] = (bf16)v;
}

// ---------------------------------------------------------------------------
// 3) Build combined samples0 (keys|labels|negs) and hop-1 samples1 ids.
// ---------------------------------------------------------------------------
__global__ __launch_bounds__(128) void samples_kernel(
    const int* __restrict__ keys, const int* __restrict__ labels,
    const int* __restrict__ negs, const int* __restrict__ adj,
    int* __restrict__ s0, int* __restrict__ s1) {
  const int t = blockIdx.x * blockDim.x + threadIdx.x;
  if (t >= S0TOT) return;
  int node;
  if (t < BATCH)            node = keys[t];
  else if (t < 2 * BATCH)   node = labels[t - BATCH];
  else                      node = negs[t - 2 * BATCH];
  s0[t] = node;
  const int* a = adj + (size_t)node * (KNB * 2);
#pragma unroll
  for (int k = 0; k < KNB; ++k) s1[t * KNB + k] = a[2 * k + 1];
}

// ---------------------------------------------------------------------------
// 4) Layer-0 gather: X[row, 0:128]  = bf16( (1/K) sum_k AMP1[p_k] * feat[nb_k] )
//                    X[row,128:256] = bf16( feat[node] )
//    One wave per row; feature table (51 MB) is L2-resident.
// ---------------------------------------------------------------------------
__global__ __launch_bounds__(256) void gather_l0_kernel(
    const int* __restrict__ nodes, int nrows, const int* __restrict__ adj,
    const float* __restrict__ feat, const float* __restrict__ amp1,
    bf16* __restrict__ X) {
  const int row = blockIdx.x * 8 + (threadIdx.x >> 5);
  if (row >= nrows) return;
  const int lane = threadIdx.x & 31;
  const int node = nodes[row];
  const int* a = adj + (size_t)node * (KNB * 2);
  float4 acc = make_float4(0.f, 0.f, 0.f, 0.f);
#pragma unroll 4
  for (int k = 0; k < KNB; ++k) {
    const int p  = a[2 * k];
    const int nb = a[2 * k + 1];
    const float4 am = ((const float4*)(amp1 + (size_t)p * FDIM))[lane];
    const float4 fv = ((const float4*)(feat + (size_t)nb * FDIM))[lane];
    acc.x = fmaf(am.x, fv.x, acc.x);
    acc.y = fmaf(am.y, fv.y, acc.y);
    acc.z = fmaf(am.z, fv.z, acc.z);
    acc.w = fmaf(am.w, fv.w, acc.w);
  }
  const float s = 1.0f / (float)KNB;
  bf16* xr = X + (size_t)row * 256 + 4 * lane;
  bf16x4 o;
  o[0] = (bf16)(acc.x * s); o[1] = (bf16)(acc.y * s);
  o[2] = (bf16)(acc.z * s); o[3] = (bf16)(acc.w * s);
  *(bf16x4*)xr = o;
  const float4 sf = ((const float4*)(feat + (size_t)node * FDIM))[lane];
  bf16x4 o2;
  o2[0] = (bf16)sf.x; o2[1] = (bf16)sf.y; o2[2] = (bf16)sf.z; o2[3] = (bf16)sf.w;
  *(bf16x4*)(xr + 128) = o2;
}

// ---------------------------------------------------------------------------
// 5) WMMA GEMM: C = act(A[M,Kd] @ Bpacked[Kd,N] + bias), bf16 in, f32 accum.
//    The whole packed B (Kd*N*2 = 128 KB) is staged into LDS per block with
//    gfx1250 async global->LDS copies (ASYNCcnt), then the fully unrolled
//    K-loop feeds v_wmma_f32_16x16x32_bf16 from ds_load_b128 (DScnt) while
//    the VMEM pipe streams the A fragments. Wave = 16x64 output strip.
// ---------------------------------------------------------------------------
template <int NKT, bool RELU_BF16>
__global__ __launch_bounds__(256) void gemm_bf16_kernel(
    const bf16* __restrict__ A, const bf16* __restrict__ Bp,
    const float* __restrict__ bias, void* __restrict__ Cout,
    int M, int N) {
  extern __shared__ char smemraw[];                // NKT*32*N*2 bytes (128 KB)
  const int tid  = threadIdx.x;
  const int lane = tid & 31;

  // ---- async-copy packed B into LDS (each thread: 16 B chunks) ----
  {
    const int bytes = NKT * 32 * N * 2;
    const unsigned ldsbase = (unsigned)(size_t)(void*)smemraw;  // LDS offset
    const char* gp = (const char*)Bp;
    for (int off = tid * 16; off < bytes; off += 256 * 16) {
      unsigned laddr = ldsbase + (unsigned)off;
      unsigned long long gaddr = (unsigned long long)(size_t)(gp + off);
      asm volatile("global_load_async_to_lds_b128 %0, %1, off"
                   :: "v"(laddr), "v"(gaddr) : "memory");
    }
    asm volatile("s_wait_asynccnt 0x0" ::: "memory");
    __syncthreads();
  }

  const int w    = tid >> 5;
  const int half = lane >> 4;
  const int l15  = lane & 15;
  const int wavesN = N >> 6;                 // waves along N (64 cols each)
  const int mWave  = w / wavesN;
  const int nWave  = w % wavesN;
  const int rowsPerBlock = (8 / wavesN) * 16;
  const long mBase = (long)blockIdx.x * rowsPerBlock + mWave * 16;
  const int  nBase = nWave * 64;
  if (mBase >= M) return;                    // wave-uniform: EXEC stays all-ones

  floatx8 c[4];
#pragma unroll
  for (int t = 0; t < 4; ++t)
#pragma unroll
    for (int v = 0; v < 8; ++v) c[t][v] = 0.0f;

  const int lda = NKT << 5;                  // Kd elements per row
  const bf16* arow = A + (mBase + l15) * (size_t)lda;
  const uint4* sB = (const uint4*)smemraw;   // fragment-packed B in LDS

#pragma unroll
  for (int kt = 0; kt < NKT; ++kt) {
    union { unsigned u[8]; bf16x16 v; } af;
    const int kb = kt * 32 + 8 * half;
#pragma unroll
    for (int j = 0; j < 8; ++j) {
      const int kk = kb + 2 * j + (j >= 4 ? 8 : 0);
      af.u[j] = *(const unsigned*)(arow + kk);
    }
#pragma unroll
    for (int t = 0; t < 4; ++t) {
      const int nt = (nBase >> 4) + t;
      union { uint4 q[2]; bf16x16 v; } bf;
      const int fi = ((nt * NKT + kt) * 32 + lane) * 2;  // uint4 index
      bf.q[0] = sB[fi];
      bf.q[1] = sB[fi + 1];
      c[t] = __builtin_amdgcn_wmma_f32_16x16x32_bf16(
          false, af.v, false, bf.v, (short)0, c[t], false, false);
    }
  }

#pragma unroll
  for (int t = 0; t < 4; ++t) {
    const int n = nBase + t * 16 + l15;
    const float bi = bias[n];
#pragma unroll
    for (int v = 0; v < 8; ++v) {
      const long m = mBase + v + 8 * half;
      float val = c[t][v] + bi;
      if (RELU_BF16) {
        val = fmaxf(val, 0.0f);
        ((bf16*)Cout)[m * (size_t)N + n] = (bf16)val;
      } else {
        ((float*)Cout)[m * (size_t)N + n] = val;
      }
    }
  }
}

// ---------------------------------------------------------------------------
// 6) Layer-1 gather: X2[row, 0:256]   = bf16( (1/K) sum_k AMP2[p_k] * H1[row*K+k] )
//                    X2[row, 256:512] = H0[row]  (bf16 copy)
// ---------------------------------------------------------------------------
__global__ __launch_bounds__(256) void gather_l1_kernel(
    const int* __restrict__ s0, const int* __restrict__ adj,
    const float* __restrict__ amp2, const bf16* __restrict__ H1,
    const bf16* __restrict__ H0, bf16* __restrict__ X2) {
  const int row = blockIdx.x * 8 + (threadIdx.x >> 5);
  if (row >= S0TOT) return;
  const int lane = threadIdx.x & 31;
  const int node = s0[row];
  const int* a = adj + (size_t)node * (KNB * 2);
  float acc[8];
#pragma unroll
  for (int e = 0; e < 8; ++e) acc[e] = 0.0f;
#pragma unroll 2
  for (int k = 0; k < KNB; ++k) {
    const int p = a[2 * k];
    const float* ar = amp2 + (size_t)p * HDIM + 8 * lane;
    const float4 a0 = *(const float4*)(ar);
    const float4 a1 = *(const float4*)(ar + 4);
    union { uint4 q; bf16 h[8]; } hv;
    hv.q = *(const uint4*)(H1 + ((size_t)row * KNB + k) * HDIM + 8 * lane);
    acc[0] = fmaf(a0.x, (float)hv.h[0], acc[0]);
    acc[1] = fmaf(a0.y, (float)hv.h[1], acc[1]);
    acc[2] = fmaf(a0.z, (float)hv.h[2], acc[2]);
    acc[3] = fmaf(a0.w, (float)hv.h[3], acc[3]);
    acc[4] = fmaf(a1.x, (float)hv.h[4], acc[4]);
    acc[5] = fmaf(a1.y, (float)hv.h[5], acc[5]);
    acc[6] = fmaf(a1.z, (float)hv.h[6], acc[6]);
    acc[7] = fmaf(a1.w, (float)hv.h[7], acc[7]);
  }
  const float s = 1.0f / (float)KNB;
  union { uint4 q; bf16 h[8]; } ov;
#pragma unroll
  for (int e = 0; e < 8; ++e) ov.h[e] = (bf16)(acc[e] * s);
  *(uint4*)(X2 + (size_t)row * 512 + 8 * lane) = ov.q;
  // self half: copy 256 bf16 of H0[row] (one uint4 per lane)
  ((uint4*)(X2 + (size_t)row * 512 + 256))[lane] =
      ((const uint4*)(H0 + (size_t)row * HDIM))[lane];
}

// ---------------------------------------------------------------------------
// 7) Row-wise L2 normalize into d_out (wave32 shuffle reduction).
// ---------------------------------------------------------------------------
__global__ __launch_bounds__(256) void l2norm_kernel(
    const float* __restrict__ in, float* __restrict__ out) {
  const int row = blockIdx.x * 8 + (threadIdx.x >> 5);
  if (row >= S0TOT) return;
  const int lane = threadIdx.x & 31;
  const float4 v = ((const float4*)(in + (size_t)row * EDIM))[lane];
  float ss = v.x * v.x + v.y * v.y + v.z * v.z + v.w * v.w;
#pragma unroll
  for (int o = 16; o >= 1; o >>= 1) ss += __shfl_xor(ss, o, 32);
  const float sc = 1.0f / fmaxf(sqrtf(ss), 1e-12f);
  float4 r;
  r.x = v.x * sc; r.y = v.y * sc; r.z = v.z * sc; r.w = v.w * sc;
  ((float4*)(out + (size_t)row * EDIM))[lane] = r;
}

// ---------------------------------------------------------------------------
// Launcher
// ---------------------------------------------------------------------------
extern "C" void kernel_launch(void* const* d_in, const int* in_sizes, int n_in,
                              void* d_out, int out_size, void* d_ws, size_t ws_size,
                              hipStream_t stream) {
  (void)in_sizes; (void)n_in; (void)out_size; (void)ws_size;
  const int*   keys   = (const int*)d_in[0];
  const int*   labels = (const int*)d_in[1];
  const int*   negs   = (const int*)d_in[2];
  const float* feat   = (const float*)d_in[3];
  const float* wemb   = (const float*)d_in[4];
  const int*   adj    = (const int*)d_in[5];
  const float* ws1    = (const float*)d_in[6];
  const float* wn1    = (const float*)d_in[7];
  const float* wp1    = (const float*)d_in[8];
  const float* bp1    = (const float*)d_in[9];
  const float* ba1    = (const float*)d_in[10];
  const float* ws2    = (const float*)d_in[11];
  const float* wn2    = (const float*)d_in[12];
  const float* wp2    = (const float*)d_in[13];
  const float* bp2    = (const float*)d_in[14];
  const float* ba2    = (const float*)d_in[15];

  char* ws = (char*)d_ws;
  int*   s0   = (int*)(ws + OFF_S0);
  int*   s1   = (int*)(ws + OFF_S1);
  float* amp1 = (float*)(ws + OFF_AMP1);
  float* amp2 = (float*)(ws + OFF_AMP2);
  bf16*  bpk1 = (bf16*)(ws + OFF_BP1);
  bf16*  bpk2 = (bf16*)(ws + OFF_BP2);
  bf16*  xa1  = (bf16*)(ws + OFF_XA1);
  bf16*  xa0  = (bf16*)(ws + OFF_XA0);
  bf16*  h1   = (bf16*)(ws + OFF_H1);
  bf16*  h0   = (bf16*)(ws + OFF_H0);
  bf16*  xa2  = (bf16*)(ws + OFF_XA2);
  float* out2 = (float*)(ws + OFF_OUT2);

  const size_t ldsB = 128 * 1024;  // full packed B per block (both layers)

  // 1) amp tables (100 blocks x (128 + 256) lanes)
  amp_kernel<<<NWALK, 384, 0, stream>>>(wemb, wp1, bp1, wp2, bp2, amp1, amp2);
  // 2) pack [wn1;ws1] (256x256) and [wn2;ws2] (512x128) into fragment layout
  pack_w_kernel<<<(256 * 256 + 255) / 256, 256, 0, stream>>>(wn1, ws1, bpk1, 128, 256);
  pack_w_kernel<<<(512 * 128 + 255) / 256, 256, 0, stream>>>(wn2, ws2, bpk2, 256, 128);
  // 3) samples
  samples_kernel<<<(S0TOT + 127) / 128, 128, 0, stream>>>(keys, labels, negs, adj, s0, s1);
  // 4) layer-0 gathers (hop1: 51200 rows; hop0: 2560 rows)
  gather_l0_kernel<<<S1TOT / 8, 256, 0, stream>>>(s1, S1TOT, adj, feat, amp1, xa1);
  gather_l0_kernel<<<S0TOT / 8, 256, 0, stream>>>(s0, S0TOT, adj, feat, amp1, xa0);
  // 5) layer-0 GEMMs: relu([NM|FS] @ [wn1;ws1] + ba1) -> bf16
  gemm_bf16_kernel<8, true><<<S1TOT / 32, 256, ldsB, stream>>>(xa1, bpk1, ba1, h1, S1TOT, 256);
  gemm_bf16_kernel<8, true><<<S0TOT / 32, 256, ldsB, stream>>>(xa0, bpk1, ba1, h0, S0TOT, 256);
  // 6) layer-1 gather
  gather_l1_kernel<<<S0TOT / 8, 256, 0, stream>>>(s0, adj, amp2, h1, h0, xa2);
  // 7) layer-1 GEMM: [NM2|H0] @ [wn2;ws2] + ba2 -> f32
  gemm_bf16_kernel<16, false><<<S0TOT / 64, 256, ldsB, stream>>>(xa2, bpk2, ba2, out2, S0TOT, 128);
  // 8) L2 normalize rows -> d_out (keys|labels|negs rows are already in order)
  l2norm_kernel<<<S0TOT / 8, 256, 0, stream>>>(out2, (float*)d_out);
}
